// Network4Test_49263274885468
// MI455X (gfx1250) — matrix operand order repaired and checked by reading, not compiled, not run on hardware
//
#include <hip/hip_runtime.h>
#include <hip/hip_bf16.h>

typedef __attribute__((ext_vector_type(16))) __bf16 v16bf;
typedef __attribute__((ext_vector_type(8)))  __bf16 v8bf;
typedef __attribute__((ext_vector_type(8)))  float  v8f;
typedef __attribute__((ext_vector_type(4)))  float  v4f;

// GCC-style int4 to match the async-copy builtin's parameter type exactly
// (diagnostic showed: 'int __vector(4) __device__ *').
typedef int v4i_g __attribute__((vector_size(16)));
typedef __attribute__((address_space(1))) v4i_g* as1_v4i;
typedef __attribute__((address_space(3))) v4i_g* as3_v4i;

#define IN_F  2048
#define HID_F 8192
#define OUT_F 2048
#define BATCH 4096

#define BM 128
#define BN 128
#define BK 64
#define LDT 72   // padded LDS row stride (elements); 144B, 16B-aligned

// ---- CDNA5 async memory->LDS copy (ASYNCcnt path), with safe fallback ----
#if defined(__HIP_DEVICE_COMPILE__) && __has_builtin(__builtin_amdgcn_global_load_async_to_lds_b128)
  #define HAVE_ASYNC_COPY 1
#else
  #define HAVE_ASYNC_COPY 0
#endif

__device__ __forceinline__ void async_copy16(const void* g, void* l) {
#if HAVE_ASYNC_COPY
    __builtin_amdgcn_global_load_async_to_lds_b128((as1_v4i)g, (as3_v4i)l, 0, 0);
#else
    (void)g; (void)l;
#endif
}

__device__ __forceinline__ void wait_async_copies() {
#if HAVE_ASYNC_COPY
  #if __has_builtin(__builtin_amdgcn_s_wait_asynccnt)
    __builtin_amdgcn_s_wait_asynccnt(0);
  #else
    asm volatile("s_wait_asynccnt 0x0" ::: "memory");
  #endif
#endif
}

union FragBF16 { v16bf v; v8bf h[2]; };

__device__ __forceinline__ unsigned short f32_to_bf16_raw(float f) {
    unsigned u = __builtin_bit_cast(unsigned, f);
    u += 0x7FFFu + ((u >> 16) & 1u);   // round-to-nearest-even
    return (unsigned short)(u >> 16);
}

__device__ __forceinline__ unsigned long long pack4_bf16(float a, float b, float c, float d) {
    return (unsigned long long)f32_to_bf16_raw(a)
         | ((unsigned long long)f32_to_bf16_raw(b) << 16)
         | ((unsigned long long)f32_to_bf16_raw(c) << 32)
         | ((unsigned long long)f32_to_bf16_raw(d) << 48);
}

// 16 WMMAs (4x2 16x16 tiles x 2 k-steps of 32) for one BK=64 stage from LDS tiles.
__device__ __forceinline__ void mma_stage(const unsigned short* lA, const unsigned short* lB,
                                          v8f acc[4][2], int wm, int wn, int lhalf, int l15) {
    #pragma unroll
    for (int ko = 0; ko < BK; ko += 32) {
        FragBF16 a[4], b[2];
        const int kbA = ko + lhalf * 8;   // A lane-half K base (ISA A layout)
        const int kbB = ko + lhalf * 8;   // B lane-half K base (same layout as A)
        #pragma unroll
        for (int i = 0; i < 4; ++i) {
            int row = wm * 64 + i * 16 + l15;
            a[i].h[0] = *(const v8bf*)&lA[row * LDT + kbA];
            a[i].h[1] = *(const v8bf*)&lA[row * LDT + kbA + 16];
        }
        #pragma unroll
        for (int j = 0; j < 2; ++j) {
            int n = wn * 32 + j * 16 + l15;
            b[j].h[0] = *(const v8bf*)&lB[n * LDT + kbB];
            b[j].h[1] = *(const v8bf*)&lB[n * LDT + kbB + 16];
        }
        #pragma unroll
        for (int i = 0; i < 4; ++i)
            #pragma unroll
            for (int j = 0; j < 2; ++j)
                acc[i][j] = __builtin_amdgcn_wmma_f32_16x16x32_bf16(
                    false, a[i].v, false, b[j].v, (short)0, acc[i][j], false, false);
    }
}

// ---------------- one-shot weight transpose + f32->bf16 convert ----------------
// W f32 [K][N] row-major  ->  Wt bf16 [N][K] row-major
__global__ __launch_bounds__(256)
void transpose_convert_bf16(const float* __restrict__ W, __bf16* __restrict__ Wt,
                            int K, int N) {
    __shared__ unsigned short t[32][34];   // 17-dword row stride (odd) -> conflict-free
    const int tid = threadIdx.x;
    const int n0  = blockIdx.x * 32;
    const int k0  = blockIdx.y * 32;
    {
        int kr = tid >> 3;
        int nc = (tid & 7) << 2;
        v4f v = *(const v4f*)(W + (k0 + kr) * N + n0 + nc);
        t[kr][nc + 0] = f32_to_bf16_raw(v.x);
        t[kr][nc + 1] = f32_to_bf16_raw(v.y);
        t[kr][nc + 2] = f32_to_bf16_raw(v.z);
        t[kr][nc + 3] = f32_to_bf16_raw(v.w);
    }
    __syncthreads();
    {
        int nr = tid >> 3;
        int kc = (tid & 7) << 2;
        unsigned long long pack =
              (unsigned long long)t[kc + 0][nr]
            | ((unsigned long long)t[kc + 1][nr] << 16)
            | ((unsigned long long)t[kc + 2][nr] << 32)
            | ((unsigned long long)t[kc + 3][nr] << 48);
        *(unsigned long long*)(Wt + (n0 + nr) * K + k0 + kc) = pack;
    }
}

// -------- GEMM1: h = relu(x @ W1 + b1); x f32 [B,IN]; W1t bf16 [HID,IN]; h bf16 [B,HID]
__global__ __launch_bounds__(256)
void mlp_gemm1_relu(const float* __restrict__ X, const __bf16* __restrict__ W1t,
                    const float* __restrict__ B1, __bf16* __restrict__ H) {
    __shared__ __align__(16) unsigned short ldsA[2][BM * LDT];
    __shared__ __align__(16) unsigned short ldsB[2][BN * LDT];

    const int tid   = threadIdx.x;
    const int lane  = tid & 31;
    const int wave  = tid >> 5;
    const int wm    = wave & 1;
    const int wn    = wave >> 1;
    const int m0    = blockIdx.y * BM;
    const int n0    = blockIdx.x * BN;
    const int lhalf = lane >> 4;
    const int l15   = lane & 15;
    const int S     = IN_F / BK;

    v8f zero = {};
    v8f acc[4][2];
    #pragma unroll
    for (int i = 0; i < 4; ++i)
        #pragma unroll
        for (int j = 0; j < 2; ++j) acc[i][j] = zero;

    v4f ra[8];
    auto loadA = [&](int k0) {              // global f32 loads -> regs
        #pragma unroll
        for (int p = 0; p < 8; ++p) {
            int e = p * 256 + tid; int row = e >> 4; int col = (e & 15) << 2;
            ra[p] = *(const v4f*)(X + (m0 + row) * IN_F + k0 + col);
        }
    };
    auto storeA = [&](int b) {              // convert + packed 8B LDS stores
        #pragma unroll
        for (int p = 0; p < 8; ++p) {
            int e = p * 256 + tid; int row = e >> 4; int col = (e & 15) << 2;
            *(unsigned long long*)&ldsA[b][row * LDT + col] =
                pack4_bf16(ra[p].x, ra[p].y, ra[p].z, ra[p].w);
        }
    };
#if HAVE_ASYNC_COPY
    auto stageB = [&](int k0, int b) {      // async memory->LDS, no VGPR round-trip
        #pragma unroll
        for (int p = 0; p < 4; ++p) {
            int e = p * 256 + tid; int row = e >> 3; int col = (e & 7) << 3;
            async_copy16(W1t + (n0 + row) * IN_F + k0 + col, &ldsB[b][row * LDT + col]);
        }
    };
#else
    v8bf rb[4];
    auto loadB = [&](int k0) {
        #pragma unroll
        for (int p = 0; p < 4; ++p) {
            int e = p * 256 + tid; int row = e >> 3; int col = (e & 7) << 3;
            rb[p] = *(const v8bf*)(W1t + (n0 + row) * IN_F + k0 + col);
        }
    };
    auto storeB = [&](int b) {
        #pragma unroll
        for (int p = 0; p < 4; ++p) {
            int e = p * 256 + tid; int row = e >> 3; int col = (e & 7) << 3;
            *(v8bf*)&ldsB[b][row * LDT + col] = rb[p];
        }
    };
#endif

    // prologue: stage 0
    loadA(0);
#if HAVE_ASYNC_COPY
    stageB(0, 0);
#else
    loadB(0);
    storeB(0);
#endif
    storeA(0);

    for (int s = 0; s < S; ++s) {
        const int b  = s & 1;
        const int k1 = (s + 1) * BK;
        wait_async_copies();
        __syncthreads();                    // stage s visible to all waves
        if (s + 1 < S) {
#if HAVE_ASYNC_COPY
            stageB(k1, 1 - b);              // fire-and-forget into other buffer
#else
            loadB(k1);
#endif
            loadA(k1);                      // global latency overlaps compute below
        }
        mma_stage(&ldsA[b][0], &ldsB[b][0], acc, wm, wn, lhalf, l15);
        if (s + 1 < S) {
            storeA(1 - b);
#if !HAVE_ASYNC_COPY
            storeB(1 - b);
#endif
        }
    }

    // epilogue: bias + relu, store bf16 h
    #pragma unroll
    for (int j = 0; j < 2; ++j) {
        int gn = n0 + wn * 32 + j * 16 + l15;
        float bias = B1[gn];
        #pragma unroll
        for (int i = 0; i < 4; ++i) {
            int mb = m0 + wm * 64 + i * 16 + lhalf * 8;
            #pragma unroll
            for (int r = 0; r < 8; ++r) {
                float v = acc[i][j][r] + bias;
                v = v > 0.0f ? v : 0.0f;
                H[(mb + r) * HID_F + gn] = __builtin_bit_cast(__bf16, f32_to_bf16_raw(v));
            }
        }
    }
}

// -------- GEMM2: out = sigmoid(h @ W2 + b2); h bf16 [B,HID]; W2t bf16 [OUT,HID]; out f32
__global__ __launch_bounds__(256)
void mlp_gemm2_sigmoid(const __bf16* __restrict__ H, const __bf16* __restrict__ W2t,
                       const float* __restrict__ B2, float* __restrict__ OUT) {
    __shared__ __align__(16) unsigned short ldsA[2][BM * LDT];
    __shared__ __align__(16) unsigned short ldsB[2][BN * LDT];

    const int tid   = threadIdx.x;
    const int lane  = tid & 31;
    const int wave  = tid >> 5;
    const int wm    = wave & 1;
    const int wn    = wave >> 1;
    const int m0    = blockIdx.y * BM;
    const int n0    = blockIdx.x * BN;
    const int lhalf = lane >> 4;
    const int l15   = lane & 15;
    const int S     = HID_F / BK;

    v8f zero = {};
    v8f acc[4][2];
    #pragma unroll
    for (int i = 0; i < 4; ++i)
        #pragma unroll
        for (int j = 0; j < 2; ++j) acc[i][j] = zero;

#if HAVE_ASYNC_COPY
    auto stageAB = [&](int k0, int b) {     // both operands: async memory->LDS
        #pragma unroll
        for (int p = 0; p < 4; ++p) {
            int e = p * 256 + tid; int row = e >> 3; int col = (e & 7) << 3;
            async_copy16(H   + (m0 + row) * HID_F + k0 + col, &ldsA[b][row * LDT + col]);
            async_copy16(W2t + (n0 + row) * HID_F + k0 + col, &ldsB[b][row * LDT + col]);
        }
    };
#else
    v8bf ra[4], rb[4];
    auto loadAB = [&](int k0) {
        #pragma unroll
        for (int p = 0; p < 4; ++p) {
            int e = p * 256 + tid; int row = e >> 3; int col = (e & 7) << 3;
            ra[p] = *(const v8bf*)(H   + (m0 + row) * HID_F + k0 + col);
            rb[p] = *(const v8bf*)(W2t + (n0 + row) * HID_F + k0 + col);
        }
    };
    auto storeAB = [&](int b) {
        #pragma unroll
        for (int p = 0; p < 4; ++p) {
            int e = p * 256 + tid; int row = e >> 3; int col = (e & 7) << 3;
            *(v8bf*)&ldsA[b][row * LDT + col] = ra[p];
            *(v8bf*)&ldsB[b][row * LDT + col] = rb[p];
        }
    };
#endif

#if HAVE_ASYNC_COPY
    stageAB(0, 0);
#else
    loadAB(0);
    storeAB(0);
#endif

    for (int s = 0; s < S; ++s) {
        const int b  = s & 1;
        const int k1 = (s + 1) * BK;
        wait_async_copies();
        __syncthreads();
        if (s + 1 < S) {
#if HAVE_ASYNC_COPY
            stageAB(k1, 1 - b);
#else
            loadAB(k1);
#endif
        }
        mma_stage(&ldsA[b][0], &ldsB[b][0], acc, wm, wn, lhalf, l15);
#if !HAVE_ASYNC_COPY
        if (s + 1 < S) storeAB(1 - b);
#endif
    }

    // epilogue: bias + sigmoid, store f32
    #pragma unroll
    for (int j = 0; j < 2; ++j) {
        int gn = n0 + wn * 32 + j * 16 + l15;
        float bias = B2[gn];
        #pragma unroll
        for (int i = 0; i < 4; ++i) {
            int mb = m0 + wm * 64 + i * 16 + lhalf * 8;
            #pragma unroll
            for (int r = 0; r < 8; ++r) {
                float z = acc[i][j][r] + bias;
                OUT[(mb + r) * OUT_F + gn] = 1.0f / (1.0f + __expf(-z));
            }
        }
    }
}

extern "C" void kernel_launch(void* const* d_in, const int* in_sizes, int n_in,
                              void* d_out, int out_size, void* d_ws, size_t ws_size,
                              hipStream_t stream) {
    const float* x  = (const float*)d_in[0];
    const float* W1 = (const float*)d_in[1];
    const float* b1 = (const float*)d_in[2];
    const float* W2 = (const float*)d_in[3];
    const float* b2 = (const float*)d_in[4];
    float* out = (float*)d_out;

    // workspace layout (exactly 128 MiB):
    //   [0, 64M)  : H   bf16 [4096][8192]
    //   [64M,96M) : W1t bf16 [8192][2048]
    //   [96M,128M): W2t bf16 [2048][8192]
    __bf16* H   = (__bf16*)d_ws;
    __bf16* W1t = H   + (size_t)BATCH * HID_F;
    __bf16* W2t = W1t + (size_t)HID_F * IN_F;

    dim3 blk(256);
    transpose_convert_bf16<<<dim3(HID_F / 32, IN_F / 32), blk, 0, stream>>>(W1, W1t, IN_F, HID_F);
    transpose_convert_bf16<<<dim3(OUT_F / 32, HID_F / 32), blk, 0, stream>>>(W2, W2t, HID_F, OUT_F);

    dim3 g1(HID_F / BN, BATCH / BM);   // 64 x 32
    dim3 g2(OUT_F / BN, BATCH / BM);   // 16 x 32
    mlp_gemm1_relu<<<g1, blk, 0, stream>>>(x, W1t, b1, H);
    mlp_gemm2_sigmoid<<<g2, blk, 0, stream>>>(H, W2t, b2, out);
}
